// LRU_17411797418611
// MI455X (gfx1250) — matrix operand order repaired and checked by reading, not compiled, not run on hardware
//
#include <hip/hip_runtime.h>
#include <hip/hip_bf16.h>

// ---------------------------------------------------------------------------
// LRU layer on MI455X (gfx1250): bf16 WMMA GEMMs (2x2 register-blocked) +
// chunked parallel scan.  T=4096 seq, H=2048 d_model, N=1024 d_hidden.
// ---------------------------------------------------------------------------

typedef __attribute__((ext_vector_type(16))) __bf16 v16bf;
typedef __attribute__((ext_vector_type(8)))  float  v8f;

constexpr int TT = 4096;   // sequence length
constexpr int HH = 2048;   // d_model
constexpr int NN = 1024;   // d_hidden
constexpr int LC = 128;    // scan chunk length
constexpr int NCH = TT / LC; // 32 chunks

union FragBF {
    v16bf v;
    uint4 q[2];
};

// ---------------------------------------------------------------------------
// Parameter prep: lambda = exp(-exp(nu) + i*exp(theta)); lambda^LC (exact,
// analytic); gamma = exp(gamma_log).
// ---------------------------------------------------------------------------
__global__ void lru_params_kernel(const float* __restrict__ nu,
                                  const float* __restrict__ theta,
                                  const float* __restrict__ gl,
                                  float* __restrict__ lamRe, float* __restrict__ lamIm,
                                  float* __restrict__ lamLRe, float* __restrict__ lamLIm,
                                  float* __restrict__ gamma) {
    int n = blockIdx.x * blockDim.x + threadIdx.x;
    if (n >= NN) return;
    float en  = expf(nu[n]);      // decay rate
    float ph  = expf(theta[n]);   // phase
    float mod = expf(-en);
    lamRe[n] = mod * cosf(ph);
    lamIm[n] = mod * sinf(ph);
    float modL = expf(-en * (float)LC);
    float phL  = ph * (float)LC;
    lamLRe[n] = modL * cosf(phL);
    lamLIm[n] = modL * sinf(phL);
    gamma[n] = expf(gl[n]);
}

// Convert inputs [T,H] f32 -> bf16
__global__ void lru_cvt_u_kernel(const float* __restrict__ u, __bf16* __restrict__ ub) {
    for (size_t i = (size_t)blockIdx.x * blockDim.x + threadIdx.x;
         i < (size_t)TT * HH; i += (size_t)gridDim.x * blockDim.x)
        ub[i] = (__bf16)u[i];
}

// B_norm = (B_re + i B_im) * gamma[n]  -> bf16, [N,H]
__global__ void lru_cvt_B_kernel(const float* __restrict__ Bre, const float* __restrict__ Bim,
                                 const float* __restrict__ gamma,
                                 __bf16* __restrict__ BreB, __bf16* __restrict__ BimB) {
    for (size_t i = (size_t)blockIdx.x * blockDim.x + threadIdx.x;
         i < (size_t)NN * HH; i += (size_t)gridDim.x * blockDim.x) {
        float g = gamma[i / HH];
        BreB[i] = (__bf16)(Bre[i] * g);
        BimB[i] = (__bf16)(Bim[i] * g);
    }
}

// C_re/C_im [H,N] f32 -> bf16
__global__ void lru_cvt_C_kernel(const float* __restrict__ Cre, const float* __restrict__ Cim,
                                 __bf16* __restrict__ CreB, __bf16* __restrict__ CimB) {
    for (size_t i = (size_t)blockIdx.x * blockDim.x + threadIdx.x;
         i < (size_t)HH * NN; i += (size_t)gridDim.x * blockDim.x) {
        CreB[i] = (__bf16)Cre[i];
        CimB[i] = (__bf16)Cim[i];
    }
}

// ---------------------------------------------------------------------------
// Fragment loaders (bf16, K-major row data).
// A-matrix 16x32 bf16 layout (ISA 7.12.2): lane m = lane&15,
//   halves 0..7  -> K = kb..kb+7, halves 8..15 -> K = kb+16..kb+23,
//   kb = k0 + 8*(lane>=16).
// B-matrix 32x16 bf16 layout: lane col = lane&15,
//   same K order as A: halves 0..7 -> K = kb..kb+7, 8..15 -> K = kb+16..kb+23.
// ---------------------------------------------------------------------------
__device__ __forceinline__ void load_fragA(FragBF& f, const __bf16* __restrict__ base,
                                           size_t row, int stride, int k0, int lane) {
    int kb = k0 + ((lane & 16) ? 8 : 0);
    const uint4* p = reinterpret_cast<const uint4*>(base + row * stride + kb);
    f.q[0] = p[0];   // K = kb .. kb+7
    f.q[1] = p[2];   // K = kb+16 .. kb+23
}

__device__ __forceinline__ void load_fragB(FragBF& f, const __bf16* __restrict__ base,
                                           size_t col, int stride, int k0, int lane) {
    int kb = k0 + ((lane & 16) ? 8 : 0);
    const uint4* p = reinterpret_cast<const uint4*>(base + col * stride + kb);
    f.q[0] = p[0];   // K = kb .. kb+7
    f.q[1] = p[2];   // K = kb+16 .. kb+23
}

#define WMMA_BF16(A, B, C) \
    __builtin_amdgcn_wmma_f32_16x16x32_bf16(false, (A).v, false, (B).v, (short)0, (C), false, false)

// ---------------------------------------------------------------------------
// GEMM1: Bu_re/Bu_im [T,N] = u_bf16 [T,H] @ Bn_{re,im} [N,H]^T  (f32 accum).
// Each wave owns a 32x32 output block (2x2 tiles, re+im = 8 accumulators);
// 8 waves/block stack along rows -> block covers 256 rows x 32 cols.
// 12 b128 loads per 8 WMMAs.
// ---------------------------------------------------------------------------
__global__ __launch_bounds__(256)
void lru_gemm_bu_kernel(const __bf16* __restrict__ U,
                        const __bf16* __restrict__ Bre, const __bf16* __restrict__ Bim,
                        float* __restrict__ BuRe, float* __restrict__ BuIm) {
    const int lane  = threadIdx.x & 31;
    const int wave  = threadIdx.x >> 5;
    const int tcol0 = blockIdx.x * 2;                  // two 16-col tiles
    const int trow0 = (blockIdx.y * 8 + wave) * 2;     // two 16-row tiles
    const size_t arow0 = (size_t)(trow0 * 16 + (lane & 15));
    const size_t arow1 = arow0 + 16;
    const size_t bcol0 = (size_t)(tcol0 * 16 + (lane & 15));
    const size_t bcol1 = bcol0 + 16;
    v8f re00 = {}, re01 = {}, re10 = {}, re11 = {};
    v8f im00 = {}, im01 = {}, im10 = {}, im11 = {};
#pragma unroll 2
    for (int k0 = 0; k0 < HH; k0 += 32) {
        FragBF a0, a1, br0, br1, bi0, bi1;
        load_fragA(a0,  U,   arow0, HH, k0, lane);
        load_fragA(a1,  U,   arow1, HH, k0, lane);
        load_fragB(br0, Bre, bcol0, HH, k0, lane);
        load_fragB(br1, Bre, bcol1, HH, k0, lane);
        load_fragB(bi0, Bim, bcol0, HH, k0, lane);
        load_fragB(bi1, Bim, bcol1, HH, k0, lane);
        re00 = WMMA_BF16(a0, br0, re00);
        re01 = WMMA_BF16(a0, br1, re01);
        re10 = WMMA_BF16(a1, br0, re10);
        re11 = WMMA_BF16(a1, br1, re11);
        im00 = WMMA_BF16(a0, bi0, im00);
        im01 = WMMA_BF16(a0, bi1, im01);
        im10 = WMMA_BF16(a1, bi0, im10);
        im11 = WMMA_BF16(a1, bi1, im11);
    }
    const int r0 = trow0 * 16 + ((lane & 16) ? 8 : 0);
    const int c0 = tcol0 * 16 + (lane & 15);
#pragma unroll
    for (int j = 0; j < 8; ++j) {
        const size_t rowA = (size_t)(r0 + j) * NN;
        const size_t rowB = (size_t)(r0 + 16 + j) * NN;
        BuRe[rowA + c0]      = re00[j];
        BuRe[rowA + c0 + 16] = re01[j];
        BuRe[rowB + c0]      = re10[j];
        BuRe[rowB + c0 + 16] = re11[j];
        BuIm[rowA + c0]      = im00[j];
        BuIm[rowA + c0 + 16] = im01[j];
        BuIm[rowB + c0]      = im10[j];
        BuIm[rowB + c0 + 16] = im11[j];
    }
}

// ---------------------------------------------------------------------------
// Scan phase 1: per (chunk, channel) local recurrence from zero state ->
// chunk-end carry b_c.
// ---------------------------------------------------------------------------
__global__ __launch_bounds__(256)
void lru_scan_carry_kernel(const float* __restrict__ BuRe, const float* __restrict__ BuIm,
                           const float* __restrict__ lamRe, const float* __restrict__ lamIm,
                           float* __restrict__ bRe, float* __restrict__ bIm) {
    const int n = blockIdx.x * blockDim.x + threadIdx.x;  // channel
    const int c = blockIdx.y;                             // chunk
    const float lre = lamRe[n], lim = lamIm[n];
    float hre = 0.f, him = 0.f;
    const int t0 = c * LC;
    for (int t = t0; t < t0 + LC; ++t) {
        const float ure = BuRe[(size_t)t * NN + n];
        const float uim = BuIm[(size_t)t * NN + n];
        const float nre = fmaf(lre, hre, fmaf(-lim, him, ure));
        const float nim = fmaf(lre, him, fmaf( lim, hre, uim));
        hre = nre; him = nim;
    }
    bRe[c * NN + n] = hre;
    bIm[c * NN + n] = him;
}

// ---------------------------------------------------------------------------
// Scan phase 2: sequential combine across NCH chunks (one 1024-thread block,
// one channel per thread). Hs[c] = state *entering* chunk c.
// ---------------------------------------------------------------------------
__global__ __launch_bounds__(1024)
void lru_scan_combine_kernel(const float* __restrict__ bRe, const float* __restrict__ bIm,
                             const float* __restrict__ lamLRe, const float* __restrict__ lamLIm,
                             float* __restrict__ HsRe, float* __restrict__ HsIm) {
    const int n = threadIdx.x;
    const float are = lamLRe[n], aim = lamLIm[n];
    float hre = 0.f, him = 0.f;
    for (int c = 0; c < NCH; ++c) {
        HsRe[c * NN + n] = hre;
        HsIm[c * NN + n] = him;
        const float cre = bRe[c * NN + n];
        const float cim = bIm[c * NN + n];
        const float nre = fmaf(are, hre, fmaf(-aim, him, cre));
        const float nim = fmaf(are, him, fmaf( aim, hre, cim));
        hre = nre; him = nim;
    }
}

// ---------------------------------------------------------------------------
// Scan phase 3: re-run recurrence per chunk from correct initial state and
// emit h_re and -h_im as bf16 (negation folded so GEMM2 is pure accumulate).
// ---------------------------------------------------------------------------
__global__ __launch_bounds__(256)
void lru_scan_apply_kernel(const float* __restrict__ BuRe, const float* __restrict__ BuIm,
                           const float* __restrict__ lamRe, const float* __restrict__ lamIm,
                           const float* __restrict__ HsRe, const float* __restrict__ HsIm,
                           __bf16* __restrict__ hreB, __bf16* __restrict__ himNegB) {
    const int n = blockIdx.x * blockDim.x + threadIdx.x;
    const int c = blockIdx.y;
    const float lre = lamRe[n], lim = lamIm[n];
    float hre = HsRe[c * NN + n];
    float him = HsIm[c * NN + n];
    const int t0 = c * LC;
    for (int t = t0; t < t0 + LC; ++t) {
        const float ure = BuRe[(size_t)t * NN + n];
        const float uim = BuIm[(size_t)t * NN + n];
        const float nre = fmaf(lre, hre, fmaf(-lim, him, ure));
        const float nim = fmaf(lre, him, fmaf( lim, hre, uim));
        hre = nre; him = nim;
        hreB[(size_t)t * NN + n]    = (__bf16)hre;
        himNegB[(size_t)t * NN + n] = (__bf16)(-him);
    }
}

// ---------------------------------------------------------------------------
// GEMM2: y[t,h] = sum_n h_re*C_re[h,n] + (-h_im)*C_im[h,n]  + D[h]*u[t,h].
// 2x2 register blocking: 16 b128 loads per 8 WMMAs; fused D*u epilogue.
// ---------------------------------------------------------------------------
__global__ __launch_bounds__(256)
void lru_gemm_y_kernel(const __bf16* __restrict__ Hre, const __bf16* __restrict__ HimNeg,
                       const __bf16* __restrict__ Cre, const __bf16* __restrict__ Cim,
                       const float* __restrict__ Dvec, const float* __restrict__ U32,
                       float* __restrict__ Y) {
    const int lane  = threadIdx.x & 31;
    const int wave  = threadIdx.x >> 5;
    const int tcol0 = blockIdx.x * 2;               // two 16-col tiles
    const int trow0 = (blockIdx.y * 8 + wave) * 2;  // two 16-row tiles
    const size_t arow0 = (size_t)(trow0 * 16 + (lane & 15));
    const size_t arow1 = arow0 + 16;
    const size_t bcol0 = (size_t)(tcol0 * 16 + (lane & 15));
    const size_t bcol1 = bcol0 + 16;
    v8f y00 = {}, y01 = {}, y10 = {}, y11 = {};
#pragma unroll 2
    for (int k0 = 0; k0 < NN; k0 += 32) {
        FragBF ar0, ar1, ai0, ai1, cr0, cr1, ci0, ci1;
        load_fragA(ar0, Hre,    arow0, NN, k0, lane);
        load_fragA(ar1, Hre,    arow1, NN, k0, lane);
        load_fragA(ai0, HimNeg, arow0, NN, k0, lane);
        load_fragA(ai1, HimNeg, arow1, NN, k0, lane);
        load_fragB(cr0, Cre,    bcol0, NN, k0, lane);
        load_fragB(cr1, Cre,    bcol1, NN, k0, lane);
        load_fragB(ci0, Cim,    bcol0, NN, k0, lane);
        load_fragB(ci1, Cim,    bcol1, NN, k0, lane);
        y00 = WMMA_BF16(ar0, cr0, y00);
        y00 = WMMA_BF16(ai0, ci0, y00);
        y01 = WMMA_BF16(ar0, cr1, y01);
        y01 = WMMA_BF16(ai0, ci1, y01);
        y10 = WMMA_BF16(ar1, cr0, y10);
        y10 = WMMA_BF16(ai1, ci0, y10);
        y11 = WMMA_BF16(ar1, cr1, y11);
        y11 = WMMA_BF16(ai1, ci1, y11);
    }
    const int r0 = trow0 * 16 + ((lane & 16) ? 8 : 0);
    const int c0 = tcol0 * 16 + (lane & 15);
    const float d0 = Dvec[c0];
    const float d1 = Dvec[c0 + 16];
#pragma unroll
    for (int j = 0; j < 8; ++j) {
        const size_t rowA = (size_t)(r0 + j) * HH;
        const size_t rowB = (size_t)(r0 + 16 + j) * HH;
        Y[rowA + c0]      = y00[j] + d0 * U32[rowA + c0];
        Y[rowA + c0 + 16] = y01[j] + d1 * U32[rowA + c0 + 16];
        Y[rowB + c0]      = y10[j] + d0 * U32[rowB + c0];
        Y[rowB + c0 + 16] = y11[j] + d1 * U32[rowB + c0 + 16];
    }
}

// ---------------------------------------------------------------------------
// Host launcher
// ---------------------------------------------------------------------------
extern "C" void kernel_launch(void* const* d_in, const int* in_sizes, int n_in,
                              void* d_out, int out_size, void* d_ws, size_t ws_size,
                              hipStream_t stream) {
    const float* inputs = (const float*)d_in[0];   // [T,H]
    const float* nu     = (const float*)d_in[1];   // [N]
    const float* theta  = (const float*)d_in[2];   // [N]
    const float* gl     = (const float*)d_in[3];   // [N]
    const float* B_re   = (const float*)d_in[4];   // [N,H]
    const float* B_im   = (const float*)d_in[5];   // [N,H]
    const float* C_re   = (const float*)d_in[6];   // [H,N]
    const float* C_im   = (const float*)d_in[7];   // [H,N]
    const float* Dv     = (const float*)d_in[8];   // [H]
    float* y = (float*)d_out;                      // [T,H]

    // Workspace carve-up (256B aligned slices).
    char* p = (char*)d_ws;
    auto carve = [&](size_t bytes) -> char* {
        char* r = p;
        p += (bytes + 255) & ~(size_t)255;
        return r;
    };
    __bf16* u_bf   = (__bf16*)carve((size_t)TT * HH * 2);   // 16 MB
    __bf16* BreB   = (__bf16*)carve((size_t)NN * HH * 2);   //  4 MB
    __bf16* BimB   = (__bf16*)carve((size_t)NN * HH * 2);   //  4 MB
    __bf16* CreB   = (__bf16*)carve((size_t)HH * NN * 2);   //  4 MB
    __bf16* CimB   = (__bf16*)carve((size_t)HH * NN * 2);   //  4 MB
    float*  BuRe   = (float*)carve((size_t)TT * NN * 4);    // 16 MB
    float*  BuIm   = (float*)carve((size_t)TT * NN * 4);    // 16 MB
    __bf16* hreB   = (__bf16*)carve((size_t)TT * NN * 2);   //  8 MB
    __bf16* himNB  = (__bf16*)carve((size_t)TT * NN * 2);   //  8 MB
    float*  lamRe  = (float*)carve(NN * 4);
    float*  lamIm  = (float*)carve(NN * 4);
    float*  lamLRe = (float*)carve(NN * 4);
    float*  lamLIm = (float*)carve(NN * 4);
    float*  gamma  = (float*)carve(NN * 4);
    float*  bRe    = (float*)carve((size_t)NCH * NN * 4);
    float*  bIm    = (float*)carve((size_t)NCH * NN * 4);
    float*  HsRe   = (float*)carve((size_t)NCH * NN * 4);
    float*  HsIm   = (float*)carve((size_t)NCH * NN * 4);

    // 1) parameter prep + bf16 conversions
    lru_params_kernel<<<dim3(NN / 256), dim3(256), 0, stream>>>(
        nu, theta, gl, lamRe, lamIm, lamLRe, lamLIm, gamma);
    lru_cvt_u_kernel<<<dim3(2048), dim3(256), 0, stream>>>(inputs, u_bf);
    lru_cvt_B_kernel<<<dim3(1024), dim3(256), 0, stream>>>(B_re, B_im, gamma, BreB, BimB);
    lru_cvt_C_kernel<<<dim3(1024), dim3(256), 0, stream>>>(C_re, C_im, CreB, CimB);

    // 2) Bu = u @ B_norm^T  (WMMA bf16, f32 accum; 32x32 per wave)
    lru_gemm_bu_kernel<<<dim3(NN / 32, TT / 256), dim3(256), 0, stream>>>(
        u_bf, BreB, BimB, BuRe, BuIm);

    // 3) chunked diagonal complex scan
    lru_scan_carry_kernel<<<dim3(NN / 256, NCH), dim3(256), 0, stream>>>(
        BuRe, BuIm, lamRe, lamIm, bRe, bIm);
    lru_scan_combine_kernel<<<dim3(1), dim3(NN), 0, stream>>>(
        bRe, bIm, lamLRe, lamLIm, HsRe, HsIm);
    lru_scan_apply_kernel<<<dim3(NN / 256, NCH), dim3(256), 0, stream>>>(
        BuRe, BuIm, lamRe, lamIm, HsRe, HsIm, hreB, himNB);

    // 4) y = Re(h @ C^T) + D*u  (WMMA bf16, f32 accum, fused epilogue)
    lru_gemm_y_kernel<<<dim3(HH / 32, TT / 256), dim3(256), 0, stream>>>(
        hreB, himNB, CreB, CimB, Dv, inputs, y);
}
